// ClusteringLayer_40295383171687
// MI455X (gfx1250) — matrix operand order repaired and checked by reading, not compiled, not run on hardware
//
#include <hip/hip_runtime.h>

typedef __attribute__((ext_vector_type(16))) __bf16 v16bf;
typedef __attribute__((ext_vector_type(8)))  float  v8f;
typedef __attribute__((ext_vector_type(4)))  __bf16 v4bf;

#define N_ROWS 131072
#define DIMS   512
#define KCL    256
#define SHRINK 0.1f

// ---------------- Kernel 0: prep (convert clusters to bf16 hi/lo, c_sq, zero accumulators) ---
__global__ __launch_bounds__(256)
void prep_kernel(const float* __restrict__ clusters,
                 __bf16* __restrict__ chi, __bf16* __restrict__ clo,
                 float* __restrict__ csq, float* __restrict__ colsum_q,
                 float* __restrict__ wqsum, float* __restrict__ S)
{
    __shared__ float part[8];
    const int k = blockIdx.x;
    const int t = threadIdx.x;
    float s = 0.f;
#pragma unroll
    for (int i = 0; i < 2; ++i) {
        int d = t + 256 * i;
        float v = clusters[k * DIMS + d];
        __bf16 h = (__bf16)v;
        chi[k * DIMS + d] = h;
        clo[k * DIMS + d] = (__bf16)(v - (float)h);
        s += v * v;
        S[k * DIMS + d] = 0.f;          // zero centroid accumulator (part of d_out)
    }
#pragma unroll
    for (int m = 1; m < 32; m <<= 1) s += __shfl_xor(s, m);
    if ((t & 31) == 0) part[t >> 5] = s;
    __syncthreads();
    if (t == 0) {
        float tot = 0.f;
#pragma unroll
        for (int w = 0; w < 8; ++w) tot += part[w];
        csq[k] = tot;
        colsum_q[k] = 0.f;
        wqsum[k] = 0.f;
    }
}

// ---------------- Kernel 1: GEMM (bf16x3 split) + soft assignment q --------------------------
__global__ __launch_bounds__(256)
void gemm_q_kernel(const float* __restrict__ x,
                   const __bf16* __restrict__ chi, const __bf16* __restrict__ clo,
                   const float* __restrict__ csq,
                   float* __restrict__ q, float* __restrict__ colsum_q)
{
    __shared__ __bf16 Ah[128][32];
    __shared__ __bf16 Al[128][32];
    __shared__ __bf16 Bh[256][32];
    __shared__ __bf16 Bl[256][32];
    __shared__ float xsqLds[128];
    __shared__ float csqLds[256];
    __shared__ float colLds[256];

    const int tid  = threadIdx.x;
    const int lane = tid & 31;
    const int wave = tid >> 5;
    const int mrow = lane & 15;
    const int half = lane >> 4;
    const size_t rowBase = (size_t)blockIdx.x * 128;

    csqLds[tid] = csq[tid];
    colLds[tid] = 0.f;
    if (tid < 128) xsqLds[tid] = 0.f;

    const v8f vzero = {0.f,0.f,0.f,0.f,0.f,0.f,0.f,0.f};
    v8f acc[16];
#pragma unroll
    for (int j = 0; j < 16; ++j) acc[j] = vzero;

    float xsqp[4] = {0.f, 0.f, 0.f, 0.f};

    // LDS byte addresses for the async B-tile copies (per-thread row of 64B).
    const unsigned bh_lds = (unsigned)(uintptr_t)&Bh[tid][0];
    const unsigned bl_lds = (unsigned)(uintptr_t)&Bl[tid][0];

    for (int dc = 0; dc < DIMS; dc += 32) {
        // Async-stage B tile (bf16 clusters in ws -> LDS, bypassing VGPRs).
        // GVS mode: saddr = uniform base, vaddr = per-lane 32-bit byte offset.
        {
            unsigned goff = (unsigned)tid * (DIMS * 2) + (unsigned)dc * 2;
#pragma unroll
            for (int i = 0; i < 4; ++i) {
                asm volatile("global_load_async_to_lds_b128 %0, %1, %2"
                             :: "v"(bh_lds + 16u * i), "v"(goff + 16u * i), "s"(chi)
                             : "memory");
                asm volatile("global_load_async_to_lds_b128 %0, %1, %2"
                             :: "v"(bl_lds + 16u * i), "v"(goff + 16u * i), "s"(clo)
                             : "memory");
            }
        }
        // Stage A tile (128x32 f32 -> bf16 hi/lo), accumulate exact ||x||^2 partials.
        // This VALU conversion work overlaps with the in-flight async B copies.
#pragma unroll
        for (int i = 0; i < 4; ++i) {
            int li   = tid + 256 * i;
            int row  = li >> 3;
            int quad = li & 7;
            const float4 v = *(const float4*)(x + (rowBase + row) * DIMS + dc + quad * 4);
            xsqp[i] += v.x * v.x + v.y * v.y + v.z * v.z + v.w * v.w;
            __bf16 h0 = (__bf16)v.x, h1 = (__bf16)v.y, h2 = (__bf16)v.z, h3 = (__bf16)v.w;
            v4bf hv = {h0, h1, h2, h3};
            v4bf lv = {(__bf16)(v.x - (float)h0), (__bf16)(v.y - (float)h1),
                       (__bf16)(v.z - (float)h2), (__bf16)(v.w - (float)h3)};
            *(v4bf*)&Ah[row][quad * 4] = hv;
            *(v4bf*)&Al[row][quad * 4] = lv;
        }
        asm volatile("s_wait_asynccnt 0x0" ::: "memory");
        __syncthreads();

        // A fragments: lane l -> row M=l%16; lanes<16 hold K {0..7,16..23}, lanes>=16 {8..15,24..31}
        v16bf ah, al;
        {
            const __bf16* ap = &Ah[wave * 16 + mrow][half * 8];
            ((uint4*)&ah)[0] = *(const uint4*)(ap);
            ((uint4*)&ah)[1] = *(const uint4*)(ap + 16);
            const __bf16* lp = &Al[wave * 16 + mrow][half * 8];
            ((uint4*)&al)[0] = *(const uint4*)(lp);
            ((uint4*)&al)[1] = *(const uint4*)(lp + 16);
        }
#pragma unroll
        for (int j = 0; j < 16; ++j) {
            // B fragments: lane l -> col N=l%16; lanes<16 hold K {0..7,16..23}, lanes>=16 {8..15,24..31}
            v16bf bh, bl;
            const __bf16* bp = &Bh[j * 16 + mrow][half * 8];
            ((uint4*)&bh)[0] = *(const uint4*)(bp);
            ((uint4*)&bh)[1] = *(const uint4*)(bp + 16);
            const __bf16* blp = &Bl[j * 16 + mrow][half * 8];
            ((uint4*)&bl)[0] = *(const uint4*)(blp);
            ((uint4*)&bl)[1] = *(const uint4*)(blp + 16);
            acc[j] = __builtin_amdgcn_wmma_f32_16x16x32_bf16(false, ah, false, bh, (short)0, acc[j], false, false);
            acc[j] = __builtin_amdgcn_wmma_f32_16x16x32_bf16(false, ah, false, bl, (short)0, acc[j], false, false);
            acc[j] = __builtin_amdgcn_wmma_f32_16x16x32_bf16(false, al, false, bh, (short)0, acc[j], false, false);
        }
        __syncthreads();
    }

    // Reduce ||x||^2 partials per row (thread t touched rows t/8 + 32*i).
#pragma unroll
    for (int i = 0; i < 4; ++i) atomicAdd(&xsqLds[(tid >> 3) + 32 * i], xsqp[i]);
    __syncthreads();

    // Finalize: d2 -> q_unnorm -> row-normalize -> store q; accumulate column sums.
    float xs[8], rs[8];
#pragma unroll
    for (int r = 0; r < 8; ++r) {
        xs[r] = xsqLds[wave * 16 + half * 8 + r];
        rs[r] = 0.f;
    }
#pragma unroll
    for (int j = 0; j < 16; ++j) {
        float cs = csqLds[j * 16 + mrow];
#pragma unroll
        for (int r = 0; r < 8; ++r) {
            float d2 = xs[r] + cs - 2.f * acc[j][r];
            d2 = fmaxf(d2, 0.f);
            float qn = 1.f / (1.f + d2);
            acc[j][r] = qn;
            rs[r] += qn;
        }
    }
#pragma unroll
    for (int r = 0; r < 8; ++r) {
        rs[r] += __shfl_xor(rs[r], 1);
        rs[r] += __shfl_xor(rs[r], 2);
        rs[r] += __shfl_xor(rs[r], 4);
        rs[r] += __shfl_xor(rs[r], 8);
        rs[r] = 1.f / rs[r];
    }
#pragma unroll
    for (int j = 0; j < 16; ++j) {
        const int kcol = j * 16 + mrow;
        float cp = 0.f;
#pragma unroll
        for (int r = 0; r < 8; ++r) {
            float qv = acc[j][r] * rs[r];
            cp += qv;
            size_t n = rowBase + (size_t)(wave * 16 + half * 8 + r);
            q[n * KCL + kcol] = qv;
        }
        cp += __shfl_xor(cp, 16);
        if (half == 0) atomicAdd(&colLds[kcol], cp);
    }
    __syncthreads();
    atomicAdd(&colsum_q[tid], colLds[tid]);
}

// ---------------- Kernel 2: sharpen + argmax + shrinkage (one wave per row) ------------------
__global__ __launch_bounds__(256)
void assign_kernel(const float* __restrict__ q, const float* __restrict__ colsum_q,
                   float* __restrict__ val, int* __restrict__ idxArr,
                   float* __restrict__ wqsum)
{
    __shared__ float cs[256];
    const int tid = threadIdx.x;
    cs[tid] = colsum_q[tid];
    __syncthreads();
    const int lane = tid & 31;
    const int wave = tid >> 5;
    const size_t n = (size_t)blockIdx.x * 8 + wave;
    const float* row = q + n * KCL;
    float rsum = 0.f, tmax = -1.f;
    int kmax = 0;
#pragma unroll
    for (int i = 0; i < 8; ++i) {
        int k = lane + 32 * i;
        float qv = row[k];
        float tv = qv * qv / cs[k];
        rsum += tv;
        if (tv > tmax) { tmax = tv; kmax = k; }
    }
#pragma unroll
    for (int m = 1; m < 32; m <<= 1) {
        rsum += __shfl_xor(rsum, m);
        float om = __shfl_xor(tmax, m);
        int   ok = __shfl_xor(kmax, m);
        if (om > tmax || (om == tmax && ok < kmax)) { tmax = om; kmax = ok; }
    }
    if (lane == 0) {
        float v = tmax / rsum;                 // normalized value at argmax
        float qf = (v > SHRINK) ? v : 0.f;     // relu(v-sh) + sign*sh collapses to this
        val[n] = qf;
        idxArr[n] = kmax;
        if (qf > 0.f) atomicAdd(&wqsum[kmax], qf);
    }
}

// ---------------- Kernel 3: scatter-add val[n] * x[n] into centroid accumulators -------------
__global__ __launch_bounds__(256)
void scatter_kernel(const float* __restrict__ x, const float* __restrict__ val,
                    const int* __restrict__ idxArr, float* __restrict__ S)
{
    const int tid = threadIdx.x;
    const size_t base = (size_t)blockIdx.x * 8;
    for (int rr = 0; rr < 8; ++rr) {
        size_t n = base + rr;
        float v = val[n];
        if (v > 0.f) {
            int k = idxArr[n];
            const float* xr = x + n * DIMS;
            float* sr = S + (size_t)k * DIMS;
            atomicAdd(&sr[tid],       v * xr[tid]);
            atomicAdd(&sr[tid + 256], v * xr[tid + 256]);
        }
    }
}

// ---------------- Kernel 4: normalize centroids --------------------------------------------
__global__ __launch_bounds__(256)
void norm_kernel(float* __restrict__ S, const float* __restrict__ wqsum)
{
    int i = blockIdx.x * 256 + threadIdx.x;   // 256*512 elements
    int k = i >> 9;
    float s = wqsum[k];
    S[i] = (s > 0.f) ? (S[i] / s) : 0.f;
}

// ---------------- Host launcher -------------------------------------------------------------
extern "C" void kernel_launch(void* const* d_in, const int* in_sizes, int n_in,
                              void* d_out, int out_size, void* d_ws, size_t ws_size,
                              hipStream_t stream)
{
    const float* x        = (const float*)d_in[0];   // [131072, 512]
    const float* clusters = (const float*)d_in[1];   // [256, 512]

    float* q    = (float*)d_out;                               // [N, 256]
    float* newc = (float*)d_out + (size_t)N_ROWS * KCL;        // [256, 512]

    // Workspace layout
    float* csq      = (float*)d_ws;              // 256
    float* colsum_q = csq + KCL;                 // 256
    float* wqsum    = colsum_q + KCL;            // 256
    float* val      = wqsum + KCL;               // N
    int*   idxArr   = (int*)(val + N_ROWS);      // N
    __bf16* chi     = (__bf16*)(idxArr + N_ROWS);// 256*512
    __bf16* clo     = chi + (size_t)KCL * DIMS;  // 256*512

    prep_kernel<<<KCL, 256, 0, stream>>>(clusters, chi, clo, csq, colsum_q, wqsum, newc);
    gemm_q_kernel<<<N_ROWS / 128, 256, 0, stream>>>(x, chi, clo, csq, q, colsum_q);
    assign_kernel<<<N_ROWS / 8, 256, 0, stream>>>(q, colsum_q, val, idxArr, wqsum);
    scatter_kernel<<<N_ROWS / 8, 256, 0, stream>>>(x, val, idxArr, newc);
    norm_kernel<<<(KCL * DIMS) / 256, 256, 0, stream>>>(newc, wqsum);
}
